// ScoreNet_discretized_16329465660122
// MI455X (gfx1250) — matrix operand order repaired and checked by reading, not compiled, not run on hardware
//
#include <hip/hip_runtime.h>
#include <hip/hip_bf16.h>

#define Hdim 256
#define Nn   50000
#define Ee   400000
#define WV   2          // waves per block (64 threads)

typedef __bf16 bf16;
typedef bf16  bf16x4  __attribute__((ext_vector_type(4)));
typedef bf16  bf16x8  __attribute__((ext_vector_type(8)));
typedef bf16  bf16x16 __attribute__((ext_vector_type(16)));
typedef float v8f     __attribute__((ext_vector_type(8)));

// ---------------------------------------------------------------------------
// Wave-level 16xK * KxN(16) -> 16x16 f32 accumulate using v_wmma_f32_16x16x32_bf16
// A: LDS, row-major [16][K] bf16.  16-bit A 16x32 (same k order as B):
//   lane L: m=L&15, khalf=L>>4; elements e -> K = 16*khalf + e (contig)
//   => two b128 LDS loads.
// B: global, weight pre-transposed WT[n][k].  32x16 16-bit B:
//   lane L: n=L&15, khalf=L>>4; elements e -> K = 16*khalf + e (contig 32B).
// C/D: VGPR r holds row m = 8*khalf + r, col n = L&15.
// ---------------------------------------------------------------------------
template<int K>
__device__ __forceinline__ v8f wmma_strip(const bf16* As, const bf16* wrow,
                                          int lrow, int khalf) {
  v8f acc = {0.f,0.f,0.f,0.f,0.f,0.f,0.f,0.f};
  const bf16* arow = As + lrow * K + 16 * khalf;
#pragma unroll
  for (int ks = 0; ks < K / 32; ++ks) {
    bf16x8 lo = *(const bf16x8*)(arow + ks * 32);
    bf16x8 hi = *(const bf16x8*)(arow + ks * 32 + 8);
    bf16x16 a;
#pragma unroll
    for (int i = 0; i < 8; ++i) { a[i] = lo[i]; a[i + 8] = hi[i]; }
    bf16x16 b = *(const bf16x16*)(wrow + ks * 32);
    acc = __builtin_amdgcn_wmma_f32_16x16x32_bf16(false, a, false, b,
                                                  (short)0, acc, false, false);
  }
  return acc;
}

// ---------------------------------------------------------------------------
// Weight transpose + bf16 convert:  WT[n][k] = (bf16) W[k][n]
// ---------------------------------------------------------------------------
__global__ void k_wt(const float* __restrict__ W, bf16* __restrict__ WT, int K, int N) {
  int i = blockIdx.x * blockDim.x + threadIdx.x;
  if (i < K * N) {
    int k = i / N, n = i % N;
    WT[n * K + k] = (bf16)W[k * N + n];
  }
}

// h[n][c] = node_emb[node_type[n]][c]
__global__ void k_embed(const int* __restrict__ node_type,
                        const float* __restrict__ node_emb, float* __restrict__ h) {
  int i = blockIdx.x * blockDim.x + threadIdx.x;      // over N*H/4
  int node = i / (Hdim / 4), c4 = (i % (Hdim / 4)) * 4;
  float4 v = *(const float4*)(node_emb + node_type[node] * Hdim + c4);
  *(float4*)(h + node * Hdim + c4) = v;
}

// ---------------------------------------------------------------------------
// Input MLP + bond:  bond = (relu(pd*w1+b1) @ w2 + b2) * edge_emb[edge_type]
// ---------------------------------------------------------------------------
__global__ __launch_bounds__(WV * 32)
void k_bond(const float* __restrict__ d, const float* __restrict__ d_noise,
            const float* __restrict__ in_w1, const float* __restrict__ in_b1,
            const bf16* __restrict__ in_w2T, const float* __restrict__ in_b2,
            const int* __restrict__ edge_type, const float* __restrict__ edge_emb,
            bf16* __restrict__ bond) {
  __shared__ bf16 Xs[WV][16 * Hdim];
  int wave = threadIdx.x >> 5, lane = threadIdx.x & 31;
  int tile = blockIdx.x * WV + wave;
  int e0 = tile * 16;
  bf16* X = Xs[wave];
  for (int t = lane; t < 16 * (Hdim / 4); t += 32) {
    int m = t >> 6, k4 = (t & 63) * 4;
    float pd = d[e0 + m] + d_noise[e0 + m];
    float4 w = *(const float4*)(in_w1 + k4);
    float4 bb = *(const float4*)(in_b1 + k4);
    bf16x4 xv;
    xv[0] = (bf16)fmaxf(pd * w.x + bb.x, 0.f);
    xv[1] = (bf16)fmaxf(pd * w.y + bb.y, 0.f);
    xv[2] = (bf16)fmaxf(pd * w.z + bb.z, 0.f);
    xv[3] = (bf16)fmaxf(pd * w.w + bb.w, 0.f);
    *(bf16x4*)(X + m * Hdim + k4) = xv;
  }
  __syncthreads();
  int lrow = lane & 15, khalf = lane >> 4;
  for (int nt = 0; nt < Hdim / 16; ++nt) {
    int n = nt * 16 + lrow;
    v8f acc = wmma_strip<Hdim>(X, in_w2T + n * Hdim + 16 * khalf, lrow, khalf);
    float bv = in_b2[n];
#pragma unroll
    for (int r = 0; r < 8; ++r) {
      int e = e0 + 8 * khalf + r;
      float ee = edge_emb[edge_type[e] * Hdim + n];
      bond[e * Hdim + n] = (bf16)((acc[r] + bv) * ee);
    }
  }
}

// ---------------------------------------------------------------------------
// Message passing: agg[dst] += relu(h[src] + bond)    (thread = edge x 8 cols)
// ---------------------------------------------------------------------------
__global__ void k_msg(const float* __restrict__ h, const bf16* __restrict__ bond,
                      const int* __restrict__ ei, float* __restrict__ agg) {
  int i = blockIdx.x * blockDim.x + threadIdx.x;      // over E*(H/8)
  int e = i / (Hdim / 8), c0 = (i % (Hdim / 8)) * 8;
  int s = ei[e], dd = ei[Ee + e];
  const float* hs = h + s * Hdim + c0;
  bf16x8 bv = *(const bf16x8*)(bond + e * Hdim + c0);
  float4 h0 = *(const float4*)(hs);
  float4 h1 = *(const float4*)(hs + 4);
  float* ag = agg + dd * Hdim + c0;
  float v[8] = { h0.x + (float)bv[0], h0.y + (float)bv[1],
                 h0.z + (float)bv[2], h0.w + (float)bv[3],
                 h1.x + (float)bv[4], h1.y + (float)bv[5],
                 h1.z + (float)bv[6], h1.w + (float)bv[7] };
#pragma unroll
  for (int j = 0; j < 8; ++j) unsafeAtomicAdd(ag + j, fmaxf(v[j], 0.f));
}

// ---------------------------------------------------------------------------
// Fused GIN conv:  h += [relu]( relu((h+agg)@W1+b1) @ W2 + b2 )
// ---------------------------------------------------------------------------
__global__ __launch_bounds__(WV * 32)
void k_conv(float* __restrict__ h, const float* __restrict__ agg,
            const bf16* __restrict__ W1T, const float* __restrict__ b1,
            const bf16* __restrict__ W2T, const float* __restrict__ b2, int relu2) {
  __shared__ bf16 Xs[WV][16 * Hdim];
  __shared__ bf16 Ts[WV][16 * Hdim];
  int wave = threadIdx.x >> 5, lane = threadIdx.x & 31;
  int tile = blockIdx.x * WV + wave;
  bool active = tile < (Nn / 16);
  int row0 = (active ? tile : 0) * 16;
  bf16* X = Xs[wave];
  bf16* T = Ts[wave];
  const float* hrow = h + row0 * Hdim;
  const float* arow = agg + row0 * Hdim;
  for (int t = lane; t < 16 * (Hdim / 4); t += 32) {   // tile rows contiguous
    int idx = t * 4;
    float4 hv = *(const float4*)(hrow + idx);
    float4 av = *(const float4*)(arow + idx);
    bf16x4 xv;
    xv[0] = (bf16)(hv.x + av.x); xv[1] = (bf16)(hv.y + av.y);
    xv[2] = (bf16)(hv.z + av.z); xv[3] = (bf16)(hv.w + av.w);
    *(bf16x4*)(X + idx) = xv;
  }
  __syncthreads();
  int lrow = lane & 15, khalf = lane >> 4;
  for (int nt = 0; nt < Hdim / 16; ++nt) {             // GEMM1 -> relu -> T
    int n = nt * 16 + lrow;
    v8f acc = wmma_strip<Hdim>(X, W1T + n * Hdim + 16 * khalf, lrow, khalf);
    float bv = b1[n];
#pragma unroll
    for (int r = 0; r < 8; ++r)
      T[(8 * khalf + r) * Hdim + n] = (bf16)fmaxf(acc[r] + bv, 0.f);
  }
  __syncthreads();
  for (int nt = 0; nt < Hdim / 16; ++nt) {             // GEMM2 -> (+relu) + h
    int n = nt * 16 + lrow;
    v8f acc = wmma_strip<Hdim>(T, W2T + n * Hdim + 16 * khalf, lrow, khalf);
    float bv = b2[n];
    if (active) {
#pragma unroll
      for (int r = 0; r < 8; ++r) {
        float v = acc[r] + bv;
        if (relu2) v = fmaxf(v, 0.f);
        int row = row0 + 8 * khalf + r;
        h[row * Hdim + n] = h[row * Hdim + n] + v;
      }
    }
  }
}

// ---------------------------------------------------------------------------
// Fused output MLP: feat=[h_src*h_dst, bond] (2H) -> H relu -> H/2 relu -> 1
// ---------------------------------------------------------------------------
__global__ __launch_bounds__(WV * 32)
void k_out(const float* __restrict__ h, const bf16* __restrict__ bond,
           const int* __restrict__ ei, const int* __restrict__ node2graph,
           const int* __restrict__ noise_level, const float* __restrict__ sigmas,
           const bf16* __restrict__ w1T, const float* __restrict__ ob1,
           const bf16* __restrict__ w2T, const float* __restrict__ ob2,
           const float* __restrict__ w3, const float* __restrict__ ob3,
           float* __restrict__ scores) {
  __shared__ bf16  Xs[WV][16 * 2 * Hdim];
  __shared__ bf16  S1[WV][16 * Hdim];
  __shared__ float Sdot[WV][16];
  const int H2 = 2 * Hdim;
  int wave = threadIdx.x >> 5, lane = threadIdx.x & 31;
  int tile = blockIdx.x * WV + wave;
  int e0 = tile * 16;
  bf16* X = Xs[wave];
  if (lane < 16) Sdot[wave][lane] = 0.f;
  for (int t = lane; t < 16 * (Hdim / 4); t += 32) {   // cols 0..H-1: h_src*h_dst
    int m = t >> 6, c4 = (t & 63) * 4;
    int e = e0 + m, s = ei[e], dd = ei[Ee + e];
    float4 a = *(const float4*)(h + s * Hdim + c4);
    float4 b = *(const float4*)(h + dd * Hdim + c4);
    bf16x4 xv;
    xv[0] = (bf16)(a.x * b.x); xv[1] = (bf16)(a.y * b.y);
    xv[2] = (bf16)(a.z * b.z); xv[3] = (bf16)(a.w * b.w);
    *(bf16x4*)(X + m * H2 + c4) = xv;
  }
  for (int t = lane; t < 16 * (Hdim / 8); t += 32) {   // cols H..2H-1: bond
    int m = t >> 5, c8 = (t & 31) * 8;
    *(bf16x8*)(X + m * H2 + Hdim + c8) =
        *(const bf16x8*)(bond + (e0 + m) * Hdim + c8);
  }
  __syncthreads();
  int lrow = lane & 15, khalf = lane >> 4;
  bf16* T = S1[wave];
  for (int nt = 0; nt < Hdim / 16; ++nt) {             // GEMM1 (K=2H) -> relu
    int n = nt * 16 + lrow;
    v8f acc = wmma_strip<2 * Hdim>(X, w1T + n * H2 + 16 * khalf, lrow, khalf);
    float bv = ob1[n];
#pragma unroll
    for (int r = 0; r < 8; ++r)
      T[(8 * khalf + r) * Hdim + n] = (bf16)fmaxf(acc[r] + bv, 0.f);
  }
  __syncthreads();
  for (int nt = 0; nt < Hdim / 32; ++nt) {             // GEMM2 (N=H/2) -> relu -> dot w3
    int n = nt * 16 + lrow;
    v8f acc = wmma_strip<Hdim>(T, w2T + n * Hdim + 16 * khalf, lrow, khalf);
    float bv = ob2[n], wv = w3[n];
#pragma unroll
    for (int r = 0; r < 8; ++r) {
      float v = fmaxf(acc[r] + bv, 0.f) * wv;
      atomicAdd(&Sdot[wave][8 * khalf + r], v);        // ds_add_f32
    }
  }
  __syncthreads();
  if (lane < 16) {
    int e = e0 + lane;
    float sig = sigmas[noise_level[node2graph[ei[e]]]];
    scores[e] = (Sdot[wave][lane] + ob3[0]) / sig;
  }
}

// target and edge_sigmas outputs
__global__ void k_tgt(const int* __restrict__ ei, const int* __restrict__ node2graph,
                      const int* __restrict__ noise_level, const float* __restrict__ sigmas,
                      const float* __restrict__ d_noise, float* __restrict__ out) {
  int e = blockIdx.x * blockDim.x + threadIdx.x;
  if (e < Ee) {
    float sig = sigmas[noise_level[node2graph[ei[e]]]];
    out[Ee + e] = -d_noise[e] / (sig * sig);
    out[2 * Ee + e] = sig;
  }
}

// ---------------------------------------------------------------------------
extern "C" void kernel_launch(void* const* d_in, const int* in_sizes, int n_in,
                              void* d_out, int out_size, void* d_ws, size_t ws_size,
                              hipStream_t stream) {
  (void)in_sizes; (void)n_in; (void)out_size; (void)ws_size;
  const int*   node_type   = (const int*)d_in[0];
  const int*   edge_type   = (const int*)d_in[1];
  const int*   edge_index  = (const int*)d_in[2];
  const int*   node2graph  = (const int*)d_in[3];
  const int*   noise_level = (const int*)d_in[4];
  const float* d           = (const float*)d_in[5];
  const float* d_noise     = (const float*)d_in[6];
  const float* sigmas      = (const float*)d_in[7];
  const float* node_emb    = (const float*)d_in[8];
  const float* edge_emb    = (const float*)d_in[9];
  const float* in_w1  = (const float*)d_in[10];
  const float* in_b1  = (const float*)d_in[11];
  const float* in_w2  = (const float*)d_in[12];
  const float* in_b2  = (const float*)d_in[13];
  const float* gin_w1 = (const float*)d_in[14];
  const float* gin_b1 = (const float*)d_in[15];
  const float* gin_w2 = (const float*)d_in[16];
  const float* gin_b2 = (const float*)d_in[17];
  const float* out_w1 = (const float*)d_in[18];
  const float* out_b1 = (const float*)d_in[19];
  const float* out_w2 = (const float*)d_in[20];
  const float* out_b2 = (const float*)d_in[21];
  const float* out_w3 = (const float*)d_in[22];
  const float* out_b3 = (const float*)d_in[23];
  float* out = (float*)d_out;

  char* w = (char*)d_ws;
  float* h    = (float*)w; w += (size_t)Nn * Hdim * 4;
  float* agg  = (float*)w; w += (size_t)Nn * Hdim * 4;
  bf16* bond  = (bf16*)w;  w += (size_t)Ee * Hdim * 2;
  bf16* inw2T = (bf16*)w;  w += (size_t)Hdim * Hdim * 2;
  bf16* gw1T  = (bf16*)w;  w += (size_t)4 * Hdim * Hdim * 2;
  bf16* gw2T  = (bf16*)w;  w += (size_t)4 * Hdim * Hdim * 2;
  bf16* ow1T  = (bf16*)w;  w += (size_t)2 * Hdim * Hdim * 2;
  bf16* ow2T  = (bf16*)w;  w += (size_t)Hdim * (Hdim / 2) * 2;

  const int HH = Hdim * Hdim;
  k_wt<<<(HH + 255) / 256, 256, 0, stream>>>(in_w2, inw2T, Hdim, Hdim);
  for (int i = 0; i < 4; ++i) {
    k_wt<<<(HH + 255) / 256, 256, 0, stream>>>(gin_w1 + i * HH, gw1T + i * HH, Hdim, Hdim);
    k_wt<<<(HH + 255) / 256, 256, 0, stream>>>(gin_w2 + i * HH, gw2T + i * HH, Hdim, Hdim);
  }
  k_wt<<<(2 * HH + 255) / 256, 256, 0, stream>>>(out_w1, ow1T, 2 * Hdim, Hdim);
  k_wt<<<(HH / 2 + 255) / 256, 256, 0, stream>>>(out_w2, ow2T, Hdim, Hdim / 2);

  k_embed<<<Nn * Hdim / 4 / 256, 256, 0, stream>>>(node_type, node_emb, h);
  k_bond<<<(Ee / 16) / WV, WV * 32, 0, stream>>>(d, d_noise, in_w1, in_b1, inw2T,
                                                 in_b2, edge_type, edge_emb, bond);
  for (int i = 0; i < 4; ++i) {
    hipMemsetAsync(agg, 0, (size_t)Nn * Hdim * 4, stream);
    k_msg<<<Ee * (Hdim / 8) / 256, 256, 0, stream>>>(h, bond, edge_index, agg);
    k_conv<<<(Nn / 16 + WV - 1) / WV, WV * 32, 0, stream>>>(
        h, agg, gw1T + i * HH, gin_b1 + i * Hdim, gw2T + i * HH, gin_b2 + i * Hdim,
        (i < 3) ? 1 : 0);
  }
  k_out<<<(Ee / 16) / WV, WV * 32, 0, stream>>>(h, bond, edge_index, node2graph,
                                                noise_level, sigmas, ow1T, out_b1,
                                                ow2T, out_b2, out_w3, out_b3, out);
  k_tgt<<<(Ee + 255) / 256, 256, 0, stream>>>(edge_index, node2graph, noise_level,
                                              sigmas, d_noise, out);
}
